// RGCNLayer_69415261438025
// MI455X (gfx1250) — matrix operand order repaired and checked by reading, not compiled, not run on hardware
//
#include <hip/hip_runtime.h>
#include <hip/hip_bf16.h>

// ---------------------------------------------------------------------------
// RGCN layer for MI455X (gfx1250, wave32, WMMA).
// Relation-sorted edges -> dense 16-edge WMMA tiles per relation, relation
// weight staged in LDS (f16, transposed for contiguous ds_load_b128 column
// fragments). All GEMM-shaped work uses v_wmma_f32_16x16x32_f16.
// All masking is branchless (mask-multiply); hot edge loop has no guards.
// ---------------------------------------------------------------------------

typedef __attribute__((ext_vector_type(16))) _Float16 v16h;
typedef __attribute__((ext_vector_type(8)))  float    v8f;

#define NN    8000
#define EE    128000
#define DIM   64
#define R2N   400
#define NB    50

static __device__ __forceinline__ v8f wmma_f16(v16h a, v16h b, v8f c) {
  // (neg_a, A, neg_b, B, c_mod, C, reuse_a, reuse_b)
  return __builtin_amdgcn_wmma_f32_16x16x32_f16(false, a, false, b, (short)0, c,
                                                false, false);
}

// A fragment (16x32 f16): lane holds row M = lane&15; lanes 0-15 carry
// K = {kb..kb+7, 16+kb..16+kb+7}, kb = (lane>>4)*8 (ISA 7.12.2).
// Branchless: unconditional b128 loads, scaled by mask m (0.0 or 1.0).
static __device__ __forceinline__ v16h load_a_row(const float* __restrict__ row,
                                                  int kBase, int lane, float m) {
  const int kb = (lane >> 4) * 8;
  const float4 x0 = *(const float4*)(row + kBase + kb);
  const float4 x1 = *(const float4*)(row + kBase + kb + 4);
  const float4 y0 = *(const float4*)(row + kBase + 16 + kb);
  const float4 y1 = *(const float4*)(row + kBase + 16 + kb + 4);
  v16h a;
  a[0]  = (_Float16)(x0.x * m); a[1]  = (_Float16)(x0.y * m);
  a[2]  = (_Float16)(x0.z * m); a[3]  = (_Float16)(x0.w * m);
  a[4]  = (_Float16)(x1.x * m); a[5]  = (_Float16)(x1.y * m);
  a[6]  = (_Float16)(x1.z * m); a[7]  = (_Float16)(x1.w * m);
  a[8]  = (_Float16)(y0.x * m); a[9]  = (_Float16)(y0.y * m);
  a[10] = (_Float16)(y0.z * m); a[11] = (_Float16)(y0.w * m);
  a[12] = (_Float16)(y1.x * m); a[13] = (_Float16)(y1.y * m);
  a[14] = (_Float16)(y1.z * m); a[15] = (_Float16)(y1.w * m);
  return a;
}

// B fragment (32x16 f16) from LDS holding B transposed as [n][64 k] f16:
// lane n = nt*16 + (lane&15); lane holds K = {kb..kb+7, 16+kb..16+kb+7}, kb = (lane>>4)*8.
static __device__ __forceinline__ v16h load_b_lds(const _Float16* bt, int nt,
                                                  int s, int lane) {
  const int n  = nt * 16 + (lane & 15);
  const int kh = s * 32 + ((lane >> 4) << 3);
  const _Float16* p = bt + n * 64 + kh;
  v16h b;
#pragma unroll
  for (int j = 0; j < 8; ++j) { b[j] = p[j]; b[j + 8] = p[16 + j]; }
  return b;
}

// ---------------- workspace init ----------------
__global__ void k_zero(float* nei, float* indeg, int* counts, int* cursor) {
  const int i = blockIdx.x * blockDim.x + threadIdx.x;
  if (i < NN * DIM) nei[i] = 0.0f;
  if (i < NN)       indeg[i] = 0.0f;
  if (i < R2N)      { counts[i] = 0; cursor[i] = 0; }
}

// ---------------- relation histogram ----------------
__global__ void k_hist(const int* __restrict__ etype, int* counts) {
  const int e = blockIdx.x * blockDim.x + threadIdx.x;
  if (e < EE) atomicAdd(&counts[etype[e]], 1);
}

// ---------------- exclusive scan over 400 buckets ----------------
__global__ void k_scan(const int* __restrict__ counts, int* offsets) {
  if (threadIdx.x == 0) {
    int acc = 0;
    for (int r = 0; r < R2N; ++r) { offsets[r] = acc; acc += counts[r]; }
  }
}

// ---------------- build relation-sorted permutation + in-degree ----------------
__global__ void k_scatter(const int* __restrict__ etype, const int* __restrict__ dst,
                          const int* __restrict__ offsets, int* cursor,
                          int* perm, float* indeg) {
  const int e = blockIdx.x * blockDim.x + threadIdx.x;
  if (e < EE) {
    const int r = etype[e];
    const int p = atomicAdd(&cursor[r], 1);
    perm[offsets[r] + p] = e;
    atomicAdd(&indeg[dst[e]], 1.0f);  // exact: sums of 1.0f
  }
}

// ---------------- rel_weight = w_comp @ weight  ([400x50]x[50x4096], f16) -----
// relw buffer is padded to 416 rows so stores need no guard.
__global__ void __launch_bounds__(256)
k_relweight(const float* __restrict__ w_comp, const float* __restrict__ weight,
            _Float16* __restrict__ relw) {
  __shared__ __align__(32) _Float16 bt[16 * 64];   // [n_local][k], K padded 50->64
  const int nt  = blockIdx.x;                      // 256 tiles over N=4096
  const int tid = threadIdx.x;
  for (int idx = tid; idx < 16 * 64; idx += 256) {
    const int nl = idx >> 6, k = idx & 63;
    const int kc = k < NB ? k : NB - 1;
    const float mk = k < NB ? 1.0f : 0.0f;
    bt[nl * 64 + k] = (_Float16)(weight[(size_t)kc * 4096 + nt * 16 + nl] * mk);
  }
  __syncthreads();
  const int lane = tid & 31, wave = tid >> 5;
  const v16h b0 = load_b_lds(bt, 0, 0, lane);
  const v16h b1 = load_b_lds(bt, 0, 1, lane);
  const int kb = (lane >> 4) * 8;
  for (int mt = wave; mt < 26; mt += 8) {          // 26 tiles cover M=400 (pad 416)
    const int row  = mt * 16 + (lane & 15);
    const float mr = row < R2N ? 1.0f : 0.0f;
    const float* wr = w_comp + (size_t)(row < R2N ? row : R2N - 1) * NB;
    v16h a0, a1;
#pragma unroll
    for (int j = 0; j < 8; ++j) {
      // K in [0,48): always < NB=50 -> only the row mask applies.
      a0[j]     = (_Float16)(wr[kb + j]      * mr);
      a0[j + 8] = (_Float16)(wr[16 + kb + j] * mr);
      a1[j]     = (_Float16)(wr[32 + kb + j] * mr);
      const int k3   = 48 + kb + j;                // K in [48,64): may exceed NB
      const float m3 = (k3 < NB) ? mr : 0.0f;
      a1[j + 8] = (_Float16)(wr[k3 < NB ? k3 : NB - 1] * m3);
    }
    v8f c = {};
    c = wmma_f16(a0, b0, c);
    c = wmma_f16(a1, b1, c);
    const int n = nt * 16 + (lane & 15);
    const int mbase = (lane >> 4) * 8;
#pragma unroll
    for (int v = 0; v < 8; ++v)                    // rows <= 415, buffer padded
      relw[(size_t)(mt * 16 + mbase + v) * 4096 + n] = (_Float16)c[v];
  }
}

// ---------------- per-relation edge gather-MM + dst scatter ----------------
__global__ void __launch_bounds__(256)
k_edge_gemm(const float* __restrict__ feat, const _Float16* __restrict__ relw,
            const int* __restrict__ src, const int* __restrict__ dst,
            const int* __restrict__ perm, const int* __restrict__ offsets,
            const int* __restrict__ counts, float* __restrict__ nei) {
  __shared__ __align__(32) _Float16 bt[64 * 64];   // relation weight, [n][k] f16
  const int r = blockIdx.x;
  const int tid = threadIdx.x;
  const _Float16* wr = relw + (size_t)r * 4096;
  for (int idx = tid; idx < 4096; idx += 256) {
    const int k = idx >> 6, n = idx & 63;
    bt[n * 64 + k] = wr[idx];
  }
  __syncthreads();
  const int lane = tid & 31, wave = tid >> 5;
  v16h bf[4][2];                                    // all B frags, reused per tile
#pragma unroll
  for (int nt = 0; nt < 4; ++nt)
#pragma unroll
    for (int s = 0; s < 2; ++s) bf[nt][s] = load_b_lds(bt, nt, s, lane);

  const int off = offsets[r], cnt = counts[r];
  const int ftiles = cnt >> 4;                      // full 16-edge tiles
  const int rem    = cnt & 15;
  const int mrow = lane & 15, mbase = (lane >> 4) * 8;

  // Hot loop: no validity checks at all.
  for (int tile = wave; tile < ftiles; tile += 8) {
    const int base = off + tile * 16;
    if (tile + 8 < ftiles) __builtin_prefetch(perm + base + 128, 0, 1);
    const int srow = src[perm[base + mrow]];
    const float* fr = feat + (size_t)srow * DIM;
    const v16h a0 = load_a_row(fr, 0,  lane, 1.0f);
    const v16h a1 = load_a_row(fr, 32, lane, 1.0f);
    v8f c[4] = {};
#pragma unroll
    for (int nt = 0; nt < 4; ++nt) {
      c[nt] = wmma_f16(a0, bf[nt][0], c[nt]);
      c[nt] = wmma_f16(a1, bf[nt][1], c[nt]);
    }
#pragma unroll
    for (int v = 0; v < 8; ++v) {                   // D: lane=N, vgpr v -> M
      float* orow = nei + (size_t)dst[perm[base + mbase + v]] * DIM;
#pragma unroll
      for (int nt = 0; nt < 4; ++nt)
        atomicAdd(&orow[nt * 16 + mrow], c[nt][v]);
    }
  }

  // Tail tile (at most one per relation), masked path.
  if (rem && wave == (ftiles & 7)) {
    const int base = off + ftiles * 16;
    const float mv = (mrow < rem) ? 1.0f : 0.0f;
    const int srow = src[perm[base + (mrow < rem ? mrow : 0)]];
    const float* fr = feat + (size_t)srow * DIM;
    const v16h a0 = load_a_row(fr, 0,  lane, mv);
    const v16h a1 = load_a_row(fr, 32, lane, mv);
    v8f c[4] = {};
#pragma unroll
    for (int nt = 0; nt < 4; ++nt) {
      c[nt] = wmma_f16(a0, bf[nt][0], c[nt]);
      c[nt] = wmma_f16(a1, bf[nt][1], c[nt]);
    }
#pragma unroll
    for (int v = 0; v < 8; ++v) {
      if (mbase + v < rem) {
        float* orow = nei + (size_t)dst[perm[base + mbase + v]] * DIM;
#pragma unroll
        for (int nt = 0; nt < 4; ++nt)
          atomicAdd(&orow[nt * 16 + mrow], c[nt][v]);
      }
    }
  }
}

// ---------------- self-loop GEMM + mean + concat into repr ----------------
__global__ void __launch_bounds__(256)
k_finalize(const float* __restrict__ feat, const float* __restrict__ slw,
           const float* __restrict__ nei, const float* __restrict__ indeg,
           float* __restrict__ out) {
  __shared__ __align__(32) _Float16 bt[64 * 64];   // self_loop_weight [n][k] f16
  const int tid = threadIdx.x;
  for (int idx = tid; idx < 4096; idx += 256) {
    const int k = idx >> 6, n = idx & 63;
    bt[n * 64 + k] = (_Float16)slw[idx];
  }
  __syncthreads();
  const int lane = tid & 31, wave = tid >> 5;
  const int tile = blockIdx.x * 8 + wave;           // 500 tiles over N=8000
  if (tile >= NN / 16) return;
  v16h bf[4][2];
#pragma unroll
  for (int nt = 0; nt < 4; ++nt)
#pragma unroll
    for (int s = 0; s < 2; ++s) bf[nt][s] = load_b_lds(bt, nt, s, lane);
  const int row0 = tile * 16;
  const float* fr = feat + (size_t)(row0 + (lane & 15)) * DIM;
  const v16h a0 = load_a_row(fr, 0,  lane, 1.0f);
  const v16h a1 = load_a_row(fr, 32, lane, 1.0f);
  v8f c[4] = {};
#pragma unroll
  for (int nt = 0; nt < 4; ++nt) {
    c[nt] = wmma_f16(a0, bf[nt][0], c[nt]);
    c[nt] = wmma_f16(a1, bf[nt][1], c[nt]);
  }
  const int mrow = lane & 15, mbase = (lane >> 4) * 8;
#pragma unroll
  for (int v = 0; v < 8; ++v) {
    const int row = row0 + mbase + v;
    const float inv = 1.0f / fmaxf(indeg[row], 1.0f);
#pragma unroll
    for (int nt = 0; nt < 4; ++nt) {
      const int ncol = nt * 16 + mrow;
      out[(size_t)row * 128 + 64 + ncol] =
          c[nt][v] + nei[(size_t)row * DIM + ncol] * inv;
    }
  }
  // repr[:, :64] = feat (pass-through), float2-vectorized, coalesced
  for (int m = 0; m < 16; ++m) {
    const int row = row0 + m;
    *(float2*)(out + (size_t)row * 128 + lane * 2) =
        *(const float2*)(feat + (size_t)row * DIM + lane * 2);
  }
}

// ---------------- rel_emb_new = rel_emb @ W_R_w.T + b ----------------
__global__ void __launch_bounds__(128)
k_relemb(const float* __restrict__ rel_emb, const float* __restrict__ Ww,
         const float* __restrict__ Wb, float* __restrict__ out) {
  __shared__ __align__(32) _Float16 bt[64 * 64];   // B = Ww^T -> [n][k] = Ww rows
  const int tid = threadIdx.x;
  for (int idx = tid; idx < 4096; idx += 128) bt[idx] = (_Float16)Ww[idx];
  __syncthreads();
  const int lane = tid & 31, nt = tid >> 5;        // 4 waves = 4 N-tiles
  const int mt = blockIdx.x;                        // 25 tiles cover M=400 exactly
  const float* fr = rel_emb + (size_t)(mt * 16 + (lane & 15)) * DIM;
  const v16h a0 = load_a_row(fr, 0,  lane, 1.0f);
  const v16h a1 = load_a_row(fr, 32, lane, 1.0f);
  const v16h b0 = load_b_lds(bt, nt, 0, lane);
  const v16h b1 = load_b_lds(bt, nt, 1, lane);
  v8f c = {};
  c = wmma_f16(a0, b0, c);
  c = wmma_f16(a1, b1, c);
  const int n = nt * 16 + (lane & 15);
  const float bias = Wb[n];
  const int mbase = (lane >> 4) * 8;
#pragma unroll
  for (int v = 0; v < 8; ++v)
    out[(size_t)(mt * 16 + mbase + v) * DIM + n] = c[v] + bias;
}

// ---------------------------------------------------------------------------
extern "C" void kernel_launch(void* const* d_in, const int* in_sizes, int n_in,
                              void* d_out, int out_size, void* d_ws, size_t ws_size,
                              hipStream_t stream) {
  const float* feat    = (const float*)d_in[0];
  const float* rel_emb = (const float*)d_in[1];
  const float* weight  = (const float*)d_in[2];
  const float* w_comp  = (const float*)d_in[3];
  const float* slw     = (const float*)d_in[4];
  const float* Ww      = (const float*)d_in[5];
  const float* Wb      = (const float*)d_in[6];
  const int*   src     = (const int*)d_in[7];
  const int*   dst     = (const int*)d_in[8];
  const int*   etype   = (const int*)d_in[9];
  float* out = (float*)d_out;

  // workspace layout (~6.0 MB); relw padded to 416 rows (unguarded stores)
  char* ws = (char*)d_ws;
  int*      counts  = (int*)(ws + 0);                            // 400 i32
  int*      offsets = (int*)(ws + 2048);                         // 400 i32
  int*      cursor  = (int*)(ws + 4096);                         // 400 i32
  int*      perm    = (int*)(ws + 8192);                         // E i32
  float*    indeg   = (float*)(ws + 8192 + (size_t)EE * 4);      // N f32
  float*    nei     = (float*)(ws + 8192 + (size_t)EE * 4 + (size_t)NN * 4);
  _Float16* relw    = (_Float16*)(ws + 8192 + (size_t)EE * 4 + (size_t)NN * 4
                                  + (size_t)NN * DIM * 4);       // 416*4096 f16

  k_zero   <<<2000, 256, 0, stream>>>(nei, indeg, counts, cursor);
  k_hist   <<<(EE + 255) / 256, 256, 0, stream>>>(etype, counts);
  k_scan   <<<1, 32, 0, stream>>>(counts, offsets);
  k_scatter<<<(EE + 255) / 256, 256, 0, stream>>>(etype, dst, offsets, cursor,
                                                  perm, indeg);
  k_relweight<<<256, 256, 0, stream>>>(w_comp, weight, relw);
  k_edge_gemm<<<R2N, 256, 0, stream>>>(feat, relw, src, dst, perm, offsets,
                                       counts, nei);
  k_finalize <<<63, 256, 0, stream>>>(feat, slw, nei, indeg, out);
  k_relemb   <<<25, 128, 0, stream>>>(rel_emb, Ww, Wb, out + (size_t)NN * 128);
}
